// BinaryRelationNetwork_78116865180358
// MI455X (gfx1250) — matrix operand order repaired and checked by reading, not compiled, not run on hardware
//
#include <hip/hip_runtime.h>

// ---------------------------------------------------------------------------
// BinaryRelationNetwork (PointNet++ encoder x2 + relation MLP) for gfx1250.
// Strategy: fuse ball-query + gather + 3x(1x1conv+BN+ReLU) + maxpool per SA
// level into one kernel; GEMMs use v_wmma_f32_16x16x32_bf16 with activations
// resident in LDS and bf16 weights streamed from global (L2-resident).
// ---------------------------------------------------------------------------

typedef __attribute__((ext_vector_type(16))) __bf16 v16bf_t;
typedef __attribute__((ext_vector_type(8)))  float  v8f_t;

struct U32x8 { uint4 lo; uint4 hi; };   // 32 bytes, bit-castable to v16bf_t

__device__ __forceinline__ unsigned short f2bf(float f) {
    unsigned int u = __float_as_uint(f);
    u += 0x7FFFu + ((u >> 16) & 1u);          // round-to-nearest-even
    return (unsigned short)(u >> 16);
}
__device__ __forceinline__ float bf2f(unsigned short h) {
    return __uint_as_float(((unsigned int)h) << 16);
}

// ---------------------------------------------------------------------------
// Weight prep: W (cin,cout) f32 -> Wt (cout,Kpad) bf16, plus folded BN affine:
//   y = relu( (x@W) * scale + shift ),  scale=g*rsqrt(v+eps),
//   shift=(b-m)*scale+beta
// ---------------------------------------------------------------------------
__global__ __launch_bounds__(64) void prep_layer_kernel(
    const float* __restrict__ W, const float* __restrict__ b,
    const float* __restrict__ beta, const float* __restrict__ g,
    const float* __restrict__ m, const float* __restrict__ v,
    int cin, int cout, int kpad,
    unsigned short* __restrict__ wt, float* __restrict__ scale,
    float* __restrict__ shift)
{
    const int co = blockIdx.x;
    for (int k = threadIdx.x; k < kpad; k += 64)
        wt[(size_t)co * kpad + k] = (k < cin) ? f2bf(W[(size_t)k * cout + co])
                                              : (unsigned short)0;
    if (threadIdx.x == 0) {
        float s = g[co] * rsqrtf(v[co] + 1e-5f);
        scale[co] = s;
        shift[co] = (b[co] - m[co]) * s + beta[co];
    }
}

// ---------------------------------------------------------------------------
// Farthest point sampling: one block per batch, sequential over S iterations.
// Emits new_xyz (B,S,3). Tie-break = first index (matches jnp.argmax).
// ---------------------------------------------------------------------------
__global__ __launch_bounds__(256) void fps_kernel(
    const float* __restrict__ xyz, int N, int S, float* __restrict__ newXyz)
{
    __shared__ float dist[4096];
    __shared__ float rv[256];
    __shared__ int   ri[256];
    __shared__ int   farS;
    __shared__ float cc[3];
    const int b = blockIdx.x, tid = threadIdx.x;
    const float* X = xyz + (size_t)b * N * 3;
    for (int i = tid; i < N; i += 256) dist[i] = 1e10f;
    if (tid == 0) farS = 0;
    __syncthreads();
    for (int it = 0; it < S; ++it) {
        if (tid == 0) {
            int f = farS;
            cc[0] = X[f*3]; cc[1] = X[f*3+1]; cc[2] = X[f*3+2];
            float* o = newXyz + ((size_t)b * S + it) * 3;
            o[0] = cc[0]; o[1] = cc[1]; o[2] = cc[2];
        }
        __syncthreads();
        float bv = -1.0f; int bi = 0x7FFFFFFF;
        const float cx = cc[0], cy = cc[1], cz = cc[2];
        for (int i = tid; i < N; i += 256) {
            float dx = X[i*3]-cx, dy = X[i*3+1]-cy, dz = X[i*3+2]-cz;
            float d = fminf(dist[i], dx*dx + dy*dy + dz*dz);
            dist[i] = d;
            if (d > bv || (d == bv && i < bi)) { bv = d; bi = i; }
        }
        rv[tid] = bv; ri[tid] = bi;
        __syncthreads();
        for (int st = 128; st > 0; st >>= 1) {
            if (tid < st) {
                float ov = rv[tid+st]; int oi = ri[tid+st];
                if (ov > rv[tid] || (ov == rv[tid] && oi < ri[tid])) {
                    rv[tid] = ov; ri[tid] = oi;
                }
            }
            __syncthreads();
        }
        if (tid == 0) farS = ri[0];
        __syncthreads();
    }
}

// ---------------------------------------------------------------------------
// Fused SA level: ball-query + gather(rel-xyz ++ feats, bf16, K zero-padded)
// + 3x [WMMA GEMM -> BN*scale+shift -> ReLU] in LDS + maxpool over group.
// Block: 256 threads (8 waves), Tc centroids (rows = Tc*64).
// ---------------------------------------------------------------------------
__global__ __launch_bounds__(256) void sa_fused_kernel(
    const float* __restrict__ xyzPrev, const float* __restrict__ featsPrev,
    int Nprev, int Cprev,
    const float* __restrict__ newXyz, int S, float r2, int Tc,
    int K0, int C1, int C2, int C3,
    const unsigned short* __restrict__ wt0, const float* __restrict__ sc0, const float* __restrict__ sh0,
    const unsigned short* __restrict__ wt1, const float* __restrict__ sc1, const float* __restrict__ sh1,
    const unsigned short* __restrict__ wt2, const float* __restrict__ sc2, const float* __restrict__ sh2,
    float* __restrict__ featsOut, int bufStrideUS)
{
    extern __shared__ char smem[];
    const int b = blockIdx.y, sbase = blockIdx.x * Tc;
    const int tid = threadIdx.x, lane = tid & 31, wave = tid >> 5;
    const int rows = Tc * 64;

    int*            idxList = (int*)smem;                         // Tc*64
    float*          cent    = (float*)(smem + Tc * 64 * 4);       // Tc*4
    unsigned short* buf0    = (unsigned short*)(smem + Tc*64*4 + Tc*16);
    unsigned short* buf1    = buf0 + (size_t)rows * bufStrideUS;

    // centroid coords -> LDS
    for (int i = tid; i < Tc * 3; i += 256) {
        int t = i / 3, j = i - t * 3;
        cent[t*4 + j] = newXyz[((size_t)b * S + sbase + t) * 3 + j];
    }
    __syncthreads();

    // ball query: wave t collects first 64 in-radius indices (ascending order)
    if (wave < Tc) {
        const int t = wave;
        const float cx = cent[t*4], cy = cent[t*4+1], cz = cent[t*4+2];
        int base = 0;
        for (int p0 = 0; p0 < Nprev && base < 64; p0 += 32) {
            int p = p0 + lane;
            bool in = false;
            if (p < Nprev) {
                const float* q = xyzPrev + ((size_t)b * Nprev + p) * 3;
                float dx = q[0]-cx, dy = q[1]-cy, dz = q[2]-cz;
                in = (dx*dx + dy*dy + dz*dz) <= r2;
            }
            unsigned mask = __builtin_amdgcn_ballot_w32(in);
            int slot = base + __builtin_popcount(mask & ((1u << lane) - 1u));
            if (in && slot < 64) idxList[t*64 + slot] = p;
            base += __builtin_popcount(mask);
        }
        if (base < 64) {                       // pad with first in-radius idx
            int first = (base > 0) ? idxList[t*64] : 0;
            for (int s2 = base + lane; s2 < 64; s2 += 32)
                idxList[t*64 + s2] = first;
        }
    }
    __syncthreads();

    // gather: row r = (centroid t, slot) -> [rel xyz | feats | zero pad] bf16
    {
        const int pitch = K0 + 8;
        for (int e = tid; e < rows * K0; e += 256) {
            int r = e / K0, k = e - r * K0;
            int t = r >> 6;
            int p = idxList[t*64 + (r & 63)];
            float val;
            if (k < 3)
                val = xyzPrev[((size_t)b * Nprev + p) * 3 + k] - cent[t*4 + k];
            else if (k < 3 + Cprev)
                val = featsPrev[((size_t)b * Nprev + p) * Cprev + (k - 3)];
            else
                val = 0.0f;
            buf0[r * pitch + k] = f2bf(val);
        }
    }
    __syncthreads();

    unsigned short* aIn  = buf0;
    unsigned short* aOut = buf1;
    const int Ks[3] = {K0, C1, C2};
    const int Ns[3] = {C1, C2, C3};
    const unsigned short* wts[3] = {wt0, wt1, wt2};
    const float* scs[3] = {sc0, sc1, sc2};
    const float* shs[3] = {sh0, sh1, sh2};
    const int MT = rows >> 4;
    const int klanA = (lane & 16);           // A frag: K split 0/16 contiguous
    const int klanB = (lane & 16);           // B frag: K split 0/16 contiguous
    const int rsel  = (lane & 16) >> 1;      // D rows 0..7 / 8..15
    const int nl    = lane & 15;

#pragma unroll
    for (int l = 0; l < 3; ++l) {
        const int K = Ks[l], N = Ns[l];
        const int pIn = K + 8, pOut = N + 8;
        const int NT = N >> 4, KT = K >> 5;
        const unsigned short* wt = wts[l];
        for (int tile = wave; tile < MT * NT; tile += 8) {
            const int mt = tile % MT, nt = tile / MT;
            v8f_t acc = {0.f, 0.f, 0.f, 0.f, 0.f, 0.f, 0.f, 0.f};
            const int rowA = (mt << 4) + nl;
            const unsigned short* aRow = aIn + rowA * pIn;
            const unsigned short* bRow = wt + (size_t)((nt << 4) + nl) * K;
            for (int kt = 0; kt < KT; ++kt) {
                U32x8 pa, pb;
                const int ka = (kt << 5) + klanA;
                pa.lo = *(const uint4*)(aRow + ka);
                pa.hi = *(const uint4*)(aRow + ka + 8);
                const int kb = (kt << 5) + klanB;
                pb.lo = *(const uint4*)(bRow + kb);
                pb.hi = *(const uint4*)(bRow + kb + 8);
                v16bf_t av = __builtin_bit_cast(v16bf_t, pa);
                v16bf_t bv = __builtin_bit_cast(v16bf_t, pb);
                acc = __builtin_amdgcn_wmma_f32_16x16x32_bf16(
                    false, av, false, bv, (short)0, acc, false, false);
            }
            const int col = (nt << 4) + nl;
            const float s = scs[l][col], h = shs[l][col];
            const int rb = (mt << 4) + rsel;
#pragma unroll
            for (int r = 0; r < 8; ++r) {
                float y = fmaxf(acc[r] * s + h, 0.0f);
                aOut[(rb + r) * pOut + col] = f2bf(y);
            }
        }
        __syncthreads();
        unsigned short* tmp = aIn; aIn = aOut; aOut = tmp;
    }

    // maxpool over ns=64 (post-ReLU, all >= 0) -> featsOut (B,S,C3) f32
    {
        const int pitch = C3 + 8;
        for (int c = tid; c < Tc * C3; c += 256) {
            int t = c / C3, ch = c - t * C3;
            const unsigned short* pr = aIn + (t * 64) * pitch + ch;
            float mm = 0.0f;
            for (int sl = 0; sl < 64; ++sl)
                mm = fmaxf(mm, bf2f(pr[sl * pitch]));
            featsOut[((size_t)b * S + sbase + t) * C3 + ch] = mm;
        }
    }
}

// ---------------------------------------------------------------------------
// fc_layer2: feats (B,512) @ W(512,64) + BN + ReLU -> (B,64)
// ---------------------------------------------------------------------------
__global__ __launch_bounds__(64) void fc_kernel(
    const float* __restrict__ feats,
    const float* __restrict__ W, const float* __restrict__ b,
    const float* __restrict__ beta, const float* __restrict__ g,
    const float* __restrict__ m, const float* __restrict__ v,
    float* __restrict__ out)
{
    const int bi = blockIdx.x, c = threadIdx.x;
    float acc = 0.0f;
    for (int k = 0; k < 512; ++k) acc += feats[bi * 512 + k] * W[k * 64 + c];
    float y = (acc + b[c] - m[c]) * rsqrtf(v[c] + 1e-5f) * g[c] + beta[c];
    out[bi * 64 + c] = fmaxf(y, 0.0f);
}

// ---------------------------------------------------------------------------
// Relation net: [sf|df|same](129) -> 64 -> 16 -> 4 -> 1. One block per batch.
// ---------------------------------------------------------------------------
__global__ __launch_bounds__(64) void relation_kernel(
    const float* __restrict__ sf, const float* __restrict__ df,
    const float* __restrict__ sp, const float* __restrict__ dp,
    const float* W0, const float* b0, const float* be0, const float* g0, const float* m0, const float* v0,
    const float* W1, const float* b1, const float* be1, const float* g1, const float* m1, const float* v1,
    const float* W2, const float* b2, const float* be2, const float* g2, const float* m2, const float* v2,
    const float* W3, const float* b3, float* __restrict__ out)
{
    __shared__ float x[129], y0[64], y1[16], y2[4];
    const int b = blockIdx.x, t = threadIdx.x;
    x[t] = sf[b * 64 + t];
    x[64 + t] = df[b * 64 + t];
    if (t == 0) {
        bool eq = (sp[b*3] == dp[b*3]) && (sp[b*3+1] == dp[b*3+1]) &&
                  (sp[b*3+2] == dp[b*3+2]);
        x[128] = eq ? 1.0f : 0.0f;
    }
    __syncthreads();
    {
        float a = 0.0f;
        for (int k = 0; k < 129; ++k) a += x[k] * W0[k * 64 + t];
        a += b0[t];
        a = (a - m0[t]) * rsqrtf(v0[t] + 1e-5f) * g0[t] + be0[t];
        y0[t] = fmaxf(a, 0.0f);
    }
    __syncthreads();
    if (t < 16) {
        float a = 0.0f;
        for (int k = 0; k < 64; ++k) a += y0[k] * W1[k * 16 + t];
        a += b1[t];
        a = (a - m1[t]) * rsqrtf(v1[t] + 1e-5f) * g1[t] + be1[t];
        y1[t] = fmaxf(a, 0.0f);
    }
    __syncthreads();
    if (t < 4) {
        float a = 0.0f;
        for (int k = 0; k < 16; ++k) a += y1[k] * W2[k * 4 + t];
        a += b2[t];
        a = (a - m2[t]) * rsqrtf(v2[t] + 1e-5f) * g2[t] + be2[t];
        y2[t] = fmaxf(a, 0.0f);
    }
    __syncthreads();
    if (t == 0) {
        float a = 0.0f;
        for (int k = 0; k < 4; ++k) a += y2[k] * W3[k];
        out[b] = a + b3[0];
    }
}

// ---------------------------------------------------------------------------
// Host side.
// Input layout (jax pytree leaf order; dicts sorted by key: W,b,beta,g,m,v):
//   0..9  : src_pcs,dst_pcs,src_pos,dst_pos,src_rot,dst_rot,src_size,
//           dst_size,src_attr,dst_attr
//   10..15: fc   {W,b,beta,g,m,v}
//   16..35: rel[0..2]{W,b,beta,g,m,v}, rel[3]{W,b}
//   36..  : sa[L][j]{W,b,beta,g,m,v}, L=0..3, j=0..2  (base 36+(L*3+j)*6)
// ---------------------------------------------------------------------------
extern "C" void kernel_launch(void* const* d_in, const int* in_sizes, int n_in,
                              void* d_out, int out_size, void* d_ws, size_t ws_size,
                              hipStream_t stream)
{
    (void)in_sizes; (void)n_in; (void)out_size; (void)ws_size;
    const int B = 16;
    static const int   S_[4]   = {512, 128, 32, 1};
    static const int   cin0_[4]= {6, 131, 259, 515};
    static const int   K0_[4]  = {32, 160, 288, 544};
    static const int   C1_[4]  = {64, 128, 256, 512};
    static const int   C2_[4]  = {64, 128, 256, 512};
    static const int   C3_[4]  = {128, 256, 512, 512};
    static const int   Tc_[4]  = {4, 2, 1, 1};
    static const int   maxW_[4]= {128, 256, 512, 544};
    static const float R2_[4]  = {0.04f, 0.16f, 0.64f, 10.24f};

    // ---- workspace carve (all 256B aligned) ----
    size_t off = 0;
    auto alloc = [&](size_t bytes) -> char* {
        off = (off + 255) & ~(size_t)255;
        char* p = (char*)d_ws + off;
        off += bytes;
        return p;
    };
    unsigned short* wt[4][3];
    float *sc[4][3], *sh[4][3];
    for (int L = 0; L < 4; ++L)
        for (int j = 0; j < 3; ++j) {
            int kpad = (j == 0) ? K0_[L] : ((j == 1) ? C1_[L] : C2_[L]);
            int cout = (j == 0) ? C1_[L] : ((j == 1) ? C2_[L] : C3_[L]);
            wt[L][j] = (unsigned short*)alloc((size_t)cout * kpad * 2);
            sc[L][j] = (float*)alloc((size_t)cout * 4);
            sh[L][j] = (float*)alloc((size_t)cout * 4);
        }
    float* xyzL[4];
    float* featsL[4];
    for (int L = 0; L < 4; ++L) {
        xyzL[L]   = (float*)alloc((size_t)B * S_[L] * 3 * 4);
        featsL[L] = (float*)alloc((size_t)B * S_[L] * C3_[L] * 4);
    }
    float* sf = (float*)alloc((size_t)B * 64 * 4);
    float* df = (float*)alloc((size_t)B * 64 * 4);

    // ---- weight prep (fold BN, transpose, bf16, K zero-pad) ----
    for (int L = 0; L < 4; ++L)
        for (int j = 0; j < 3; ++j) {
            int cin  = (j == 0) ? cin0_[L] : ((j == 1) ? C1_[L] : C2_[L]);
            int kpad = (j == 0) ? K0_[L]   : ((j == 1) ? C1_[L] : C2_[L]);
            int cout = (j == 0) ? C1_[L]   : ((j == 1) ? C2_[L] : C3_[L]);
            int p = 36 + (L * 3 + j) * 6;
            prep_layer_kernel<<<cout, 64, 0, stream>>>(
                (const float*)d_in[p + 0], (const float*)d_in[p + 1],
                (const float*)d_in[p + 2], (const float*)d_in[p + 3],
                (const float*)d_in[p + 4], (const float*)d_in[p + 5],
                cin, cout, kpad, wt[L][j], sc[L][j], sh[L][j]);
        }

    // ---- two encoders ----
    for (int enc = 0; enc < 2; ++enc) {
        const float* xyzPrev   = (const float*)d_in[enc];   // src_pcs / dst_pcs
        const float* featsPrev = (const float*)d_in[enc];
        int Nprev = 4096, Cprev = 3;
        for (int L = 0; L < 4; ++L) {
            fps_kernel<<<B, 256, 0, stream>>>(xyzPrev, Nprev, S_[L], xyzL[L]);
            int rows = Tc_[L] * 64;
            size_t lds = (size_t)Tc_[L] * 64 * 4 + (size_t)Tc_[L] * 16 +
                         2 * (size_t)rows * (maxW_[L] + 8) * 2;
            dim3 grid(S_[L] / Tc_[L], B);
            sa_fused_kernel<<<grid, 256, lds, stream>>>(
                xyzPrev, featsPrev, Nprev, Cprev,
                xyzL[L], S_[L], R2_[L], Tc_[L],
                K0_[L], C1_[L], C2_[L], C3_[L],
                wt[L][0], sc[L][0], sh[L][0],
                wt[L][1], sc[L][1], sh[L][1],
                wt[L][2], sc[L][2], sh[L][2],
                featsL[L], maxW_[L] + 8);
            xyzPrev = xyzL[L]; featsPrev = featsL[L];
            Nprev = S_[L]; Cprev = C3_[L];
        }
        fc_kernel<<<B, 64, 0, stream>>>(
            featsL[3],
            (const float*)d_in[10], (const float*)d_in[11],
            (const float*)d_in[12], (const float*)d_in[13],
            (const float*)d_in[14], (const float*)d_in[15],
            enc ? df : sf);
    }

    // ---- relation net ----
    relation_kernel<<<B, 64, 0, stream>>>(
        sf, df, (const float*)d_in[2], (const float*)d_in[3],
        (const float*)d_in[16], (const float*)d_in[17], (const float*)d_in[18],
        (const float*)d_in[19], (const float*)d_in[20], (const float*)d_in[21],
        (const float*)d_in[22], (const float*)d_in[23], (const float*)d_in[24],
        (const float*)d_in[25], (const float*)d_in[26], (const float*)d_in[27],
        (const float*)d_in[28], (const float*)d_in[29], (const float*)d_in[30],
        (const float*)d_in[31], (const float*)d_in[32], (const float*)d_in[33],
        (const float*)d_in[34], (const float*)d_in[35],
        (float*)d_out);
}
